// AECS_9560597200985
// MI455X (gfx1250) — compile-verified, matrix operand order checked by reading, not run on hardware
//
#include <hip/hip_runtime.h>
#include <hip/hip_bf16.h>

// Problem constants (from setup_inputs)
#define B_   128
#define T_   256
#define F_   128
#define L_   64
#define H_   128
#define G4_  512   // 4*H
#define KNN_ 5

typedef __attribute__((ext_vector_type(16))) __bf16 bf16x16;
typedef __attribute__((ext_vector_type(8)))  float  f32x8;

// ---------------------------------------------------------------------------
// WMMA helpers (CDNA5 wave32, V_WMMA_F32_16X16X32_BF16)
// ---------------------------------------------------------------------------
__device__ __forceinline__ f32x8 wmma_bf16(bf16x16 a, bf16x16 b, f32x8 c) {
  // (neg_a, A, neg_b, B, c_mod, C, reuse_a, reuse_b)
  return __builtin_amdgcn_wmma_f32_16x16x32_bf16(false, a, false, b, (short)0, c,
                                                 false, false);
}

// 16-bit 16x32 A/B fragment load: lanes 0-15 hold rows (K = k0..k0+15),
// lanes 16-31 hold the same rows with K = k0+16..k0+31 (contiguous k order, same for A and B).
__device__ __forceinline__ bf16x16 load_frag(const __bf16* base, int row0, int ld,
                                             int k0) {
  int lane = threadIdx.x & 31;
  int r = lane & 15, hi = lane >> 4;
  return *reinterpret_cast<const bf16x16*>(base + (size_t)(row0 + r) * ld + k0 +
                                           hi * 16);
}

__device__ __forceinline__ float sigmoidf_(float x) {
  return 1.0f / (1.0f + __expf(-x));
}

// ---------------------------------------------------------------------------
// Weight prep: f32 (K x N) -> bf16 transposed (N x K)
// ---------------------------------------------------------------------------
__global__ void aecs_wtrans(const float* __restrict__ W, __bf16* __restrict__ WT,
                            int K, int N) {
  int i = blockIdx.x * blockDim.x + threadIdx.x;
  if (i >= K * N) return;
  int kI = i / N, nI = i % N;
  WT[(size_t)nI * K + kI] = (__bf16)W[i];
}

// ---------------------------------------------------------------------------
// KNN distance prep: AM = X*M, A2M = X*X*M, M  (row-major and time-transposed)
// ---------------------------------------------------------------------------
__global__ void aecs_prep_pdist(const float* __restrict__ X,
                                const float* __restrict__ M,
                                __bf16* AMr, __bf16* A2Mr, __bf16* Mr,
                                __bf16* AMt, __bf16* A2Mt, __bf16* Mt) {
  int row = blockIdx.x;          // b*T + t
  int b = row / T_, t = row % T_;
  int f = threadIdx.x;           // F_
  size_t o = (size_t)row * F_ + f;
  float x = X[o], m = M[o];
  float am = x * m, a2m = x * x * m;
  AMr[o] = (__bf16)am; A2Mr[o] = (__bf16)a2m; Mr[o] = (__bf16)m;
  size_t ot = ((size_t)b * F_ + f) * T_ + t;
  AMt[ot] = (__bf16)am; A2Mt[ot] = (__bf16)a2m; Mt[ot] = (__bf16)m;
}

// ---------------------------------------------------------------------------
// Masked pdist: one wave per 16x16 tile of d (B,N,N); contraction over K.
// S = A2M_i.M_j + M_i.A2M_j - 2 AM_i.AM_j ; cnt = M_i.M_j
// ---------------------------------------------------------------------------
__global__ void aecs_pdist(const __bf16* __restrict__ AM,
                           const __bf16* __restrict__ A2M,
                           const __bf16* __restrict__ Mb,
                           int N, int K, float* __restrict__ d) {
  int tj = blockIdx.x, ti = blockIdx.y, b = blockIdx.z;
  size_t ib = (size_t)b * N * K;
  const __bf16* am = AM + ib;
  const __bf16* a2m = A2M + ib;
  const __bf16* mb = Mb + ib;
  int i0 = ti * 16, j0 = tj * 16;
  f32x8 s1{}, s2{}, s3{}, cn{};
  for (int k0 = 0; k0 < K; k0 += 32) {
    bf16x16 a2m_a = load_frag(a2m, i0, K, k0);
    bf16x16 am_a  = load_frag(am,  i0, K, k0);
    bf16x16 m_a   = load_frag(mb,  i0, K, k0);
    bf16x16 a2m_b = load_frag(a2m, j0, K, k0);
    bf16x16 am_b  = load_frag(am,  j0, K, k0);
    bf16x16 m_b   = load_frag(mb,  j0, K, k0);
    s1 = wmma_bf16(a2m_a, m_b,  s1);
    s2 = wmma_bf16(am_a,  am_b, s2);
    s3 = wmma_bf16(m_a,   a2m_b, s3);
    cn = wmma_bf16(m_a,   m_b,  cn);
  }
  int lane = threadIdx.x & 31, r16 = lane & 15, hi = lane >> 4;
  for (int r = 0; r < 8; ++r) {
    int gi = i0 + r + hi * 8, gj = j0 + r16;
    float S = s1[r] + s3[r] - 2.0f * s2[r];
    float c = cn[r];
    float val = (c > 0.0f && gi != gj)
                    ? sqrtf(fmaxf(S, 0.0f) / fmaxf(c, 1.0f))
                    : __builtin_huge_valf();
    d[((size_t)b * N + gi) * N + gj] = val;
  }
}

// ---------------------------------------------------------------------------
// Top-k(5) smallest + KNN weights per row of d
// ---------------------------------------------------------------------------
__global__ void aecs_knn_topk(const float* __restrict__ d, int N, int rows,
                              int* __restrict__ idx, float* __restrict__ w) {
  int row = blockIdx.x * blockDim.x + threadIdx.x;
  if (row >= rows) return;
  const float* dr = d + (size_t)row * N;
  float bv[KNN_]; int bi[KNN_];
  for (int j = 0; j < KNN_; ++j) { bv[j] = __builtin_huge_valf(); bi[j] = -1; }
  for (int s = 0; s < N; ++s) {
    float v = dr[s];
    if (v < bv[KNN_ - 1]) {
      int p = KNN_ - 1;
      while (p > 0 && bv[p - 1] > v) { bv[p] = bv[p - 1]; bi[p] = bi[p - 1]; --p; }
      bv[p] = v; bi[p] = s;
    }
  }
  int n = 0;
  for (int j = 0; j < KNN_; ++j) if (bv[j] < 1e37f) ++n;
  int lo = (n - 1) / 2; if (lo < 0) lo = 0; if (lo > KNN_ - 1) lo = KNN_ - 1;
  int hi = n / 2;       if (hi > KNN_ - 1) hi = KNN_ - 1;
  float med = 0.5f * (bv[lo] + bv[hi]);
  float sigma = (n > 0 ? med : 1.0f) + 1e-8f;
  float ww[KNN_]; float sum = 0.f;
  for (int j = 0; j < KNN_; ++j) {
    float fin = (bv[j] < 1e37f) ? 1.f : 0.f;
    float q = bv[j] / sigma;
    ww[j] = fin * __expf(-q * q);
    sum += ww[j];
  }
  float inv = 1.0f / (sum + 1e-8f);
  for (int j = 0; j < KNN_; ++j) {
    w[(size_t)row * KNN_ + j] = ww[j] * inv;
    idx[(size_t)row * KNN_ + j] = (bv[j] < 1e37f) ? bi[j] : -1;
  }
}

// ---------------------------------------------------------------------------
// Imputation (spatial: neighbors over time rows; temporal: over features)
// ---------------------------------------------------------------------------
__global__ void aecs_impute_spatial(const float* __restrict__ X,
                                    const float* __restrict__ M,
                                    const int* __restrict__ idx,
                                    const float* __restrict__ w,
                                    float* __restrict__ Xs) {
  int row = blockIdx.x;  // b*T + t
  int b = row / T_;
  int f = threadIdx.x;
  float num = 0.f, den = 0.f, condsum = 0.f;
  for (int j = 0; j < KNN_; ++j) {
    int i = idx[(size_t)row * KNN_ + j];
    if (i < 0) continue;
    size_t o = ((size_t)b * T_ + i) * F_ + f;
    float mn = M[o];
    condsum += mn;
    float vw = w[(size_t)row * KNN_ + j] * mn;
    den += vw; num += vw * X[o];
  }
  float imp = (condsum > 0.f) ? num / (den + 1e-8f) : 0.f;
  size_t o = (size_t)row * F_ + f;
  float m = M[o];
  Xs[o] = X[o] * m + (1.f - m) * imp;
}

__global__ void aecs_impute_temporal(const float* __restrict__ X,
                                     const float* __restrict__ M,
                                     const int* __restrict__ idx,
                                     const float* __restrict__ w,
                                     float* __restrict__ Xt) {
  int b = blockIdx.x / F_;
  int jf = blockIdx.x % F_;
  int t = threadIdx.x;   // T_
  int row = b * F_ + jf;
  size_t base = ((size_t)b * T_ + t) * F_;
  float num = 0.f, den = 0.f, condsum = 0.f;
  for (int j = 0; j < KNN_; ++j) {
    int i = idx[(size_t)row * KNN_ + j];
    if (i < 0) continue;
    float mn = M[base + i];
    condsum += mn;
    float vw = w[(size_t)row * KNN_ + j] * mn;
    den += vw; num += vw * X[base + i];
  }
  float imp = (condsum > 0.f) ? num / (den + 1e-8f) : 0.f;
  float m = M[base + jf];
  Xt[base + jf] = X[base + jf] * m + (1.f - m) * imp;
}

// ---------------------------------------------------------------------------
// Encoder input: [x*m, m] -> bf16 rows of length 2F
// ---------------------------------------------------------------------------
__global__ void aecs_encin(const float* __restrict__ Xv,
                           const float* __restrict__ M,
                           __bf16* __restrict__ Ain) {
  int row = blockIdx.x;  // b*T + t
  int j = threadIdx.x;   // 2F
  size_t ob = (size_t)row * F_;
  float v = (j < F_) ? Xv[ob + j] * M[ob + j] : M[ob + j - F_];
  Ain[(size_t)row * (2 * F_) + j] = (__bf16)v;
}

// ---------------------------------------------------------------------------
// Generic bf16 GEMM: C(MxN f32) = A(MxK bf16) * BT(NxK bf16)^T + bias
// blockDim 256 = 8 waves; each wave owns one 16-row tile x four 16-col tiles
// (one A fragment feeds 4 WMMAs per k-step).
// ---------------------------------------------------------------------------
__global__ void aecs_gemm_bf16(const __bf16* __restrict__ A,
                               const __bf16* __restrict__ BT,
                               const float* __restrict__ bias,
                               float* __restrict__ C, int M, int N, int K) {
  int wave = threadIdx.x >> 5;
  int tm = blockIdx.y * 8 + wave;
  int m0 = tm * 16;
  int nb = blockIdx.x * 64;     // 4 consecutive n-tiles
  if (m0 >= M) return;
  f32x8 acc0{}, acc1{}, acc2{}, acc3{};
  for (int k0 = 0; k0 < K; k0 += 32) {
    bf16x16 a  = load_frag(A, m0, K, k0);
    bf16x16 b0 = load_frag(BT, nb +  0, K, k0);
    bf16x16 b1 = load_frag(BT, nb + 16, K, k0);
    bf16x16 b2 = load_frag(BT, nb + 32, K, k0);
    bf16x16 b3 = load_frag(BT, nb + 48, K, k0);
    acc0 = wmma_bf16(a, b0, acc0);
    acc1 = wmma_bf16(a, b1, acc1);
    acc2 = wmma_bf16(a, b2, acc2);
    acc3 = wmma_bf16(a, b3, acc3);
  }
  int lane = threadIdx.x & 31, r16 = lane & 15, hi = lane >> 4;
  f32x8 accs[4] = {acc0, acc1, acc2, acc3};
  for (int jj = 0; jj < 4; ++jj) {
    int n0 = nb + jj * 16;
    float bb = bias ? bias[n0 + r16] : 0.f;
    for (int r = 0; r < 8; ++r) {
      int m = m0 + r + hi * 8;
      C[(size_t)m * N + n0 + r16] = accs[jj][r] + bb;
    }
  }
}

// ---------------------------------------------------------------------------
// LSTM recurrence. One workgroup per 16 batch rows. WhT (4H x H, bf16) is
// DMA'd into LDS with GLOBAL_LOAD_ASYNC_TO_LDS_B128 (ASYNCcnt path) and stays
// resident (128 KB; CDNA5 WGP has 320 KB LDS). Wave w owns hidden slice
// n in [16w,16w+16); its 4 accumulators are the i/f/c/o gate tiles.
// ---------------------------------------------------------------------------
__global__ void aecs_lstm_rec(const float* __restrict__ xg,   // (B,T,4H) f32
                              const __bf16* __restrict__ WhT, // (4H,H) bf16
                              float* __restrict__ hs) {       // (B,T,H) f32
  extern __shared__ char smem[];
  __bf16* sWh = (__bf16*)smem;                           // 4H*H bf16 @ offset 0
  __bf16* sh  = (__bf16*)(smem + (size_t)G4_ * H_ * 2);  // 16*H bf16
  int tid = threadIdx.x;
  int b0 = blockIdx.x * 16;
  {
    // Async memory->LDS copy: each lane moves 16B per op, 256 threads x 32
    // iterations cover 4H*H*2 = 128 KB. Tracked by ASYNCcnt.
    const int TOTAL = G4_ * H_ * 2;            // bytes
    unsigned lds_off = (unsigned)(tid * 16);
    unsigned long long gaddr = (unsigned long long)WhT + tid * 16;
    for (int i = 0; i < TOTAL / (256 * 16); ++i) {
      asm volatile("global_load_async_to_lds_b128 %0, %1, off"
                   :: "v"(lds_off), "v"(gaddr)
                   : "memory");
      lds_off += 256 * 16;
      gaddr += 256 * 16;
    }
    for (int i = tid; i < 16 * H_; i += blockDim.x) sh[i] = (__bf16)0.0f;
    asm volatile("s_wait_asynccnt 0x0" ::: "memory");
  }
  __syncthreads();
  int wave = tid >> 5, lane = tid & 31, r16 = lane & 15, hi = lane >> 4;
  int n = wave * 16 + r16;   // hidden index (lanes 16-31 duplicate per C layout)
  float c[8];
  for (int r = 0; r < 8; ++r) c[r] = 0.f;

  for (int t = 0; t < T_; ++t) {
    f32x8 ai{}, af{}, ac{}, ao{};
    for (int k0 = 0; k0 < H_; k0 += 32) {
      bf16x16 a  = load_frag(sh, 0, H_, k0);
      bf16x16 bi = load_frag(sWh, wave * 16 + 0 * H_, H_, k0);
      bf16x16 bf = load_frag(sWh, wave * 16 + 1 * H_, H_, k0);
      bf16x16 bc = load_frag(sWh, wave * 16 + 2 * H_, H_, k0);
      bf16x16 bo = load_frag(sWh, wave * 16 + 3 * H_, H_, k0);
      ai = wmma_bf16(a, bi, ai);
      af = wmma_bf16(a, bf, af);
      ac = wmma_bf16(a, bc, ac);
      ao = wmma_bf16(a, bo, ao);
    }
    __syncthreads();   // old h fully consumed
    for (int r = 0; r < 8; ++r) {
      int m = r + hi * 8;
      size_t gb = ((size_t)(b0 + m) * T_ + t) * G4_;
      float zi = ai[r] + xg[gb + n];
      float zf = af[r] + xg[gb + H_ + n];
      float zc = ac[r] + xg[gb + 2 * H_ + n];
      float zo = ao[r] + xg[gb + 3 * H_ + n];
      float cn = sigmoidf_(zf) * c[r] + sigmoidf_(zi) * tanhf(zc);
      float hn = sigmoidf_(zo) * tanhf(cn);
      c[r] = cn;
      hs[((size_t)(b0 + m) * T_ + t) * H_ + n] = hn;
      sh[m * H_ + n] = (__bf16)hn;
    }
    __syncthreads();   // new h visible to all waves
  }
}

// ---------------------------------------------------------------------------
// BatchNorm (inference) -> bf16 activation for next GEMM
// ---------------------------------------------------------------------------
__global__ void aecs_bn_bf16(const float* __restrict__ hsin,
                             const float* __restrict__ beta,
                             const float* __restrict__ gamma,
                             const float* __restrict__ mean,
                             const float* __restrict__ var,
                             __bf16* __restrict__ out) {
  int row = blockIdx.x, cidx = threadIdx.x;  // H_
  size_t o = (size_t)row * H_ + cidx;
  float v = (hsin[o] - mean[cidx]) * rsqrtf(var[cidx] + 1e-3f) * gamma[cidx] +
            beta[cidx];
  out[o] = (__bf16)v;
}

// ---------------------------------------------------------------------------
// Gating feature vector: [mean_t z_o, mean_t z_s, mean_t z_t, rho]
// ---------------------------------------------------------------------------
__global__ void aecs_feat(const float* __restrict__ zo,
                          const float* __restrict__ zs,
                          const float* __restrict__ zt,
                          const float* __restrict__ M, float* __restrict__ feat) {
  __shared__ float sm[64];
  int b = blockIdx.x, l = threadIdx.x;  // 64
  float so = 0.f, ss = 0.f, st = 0.f;
  for (int t = 0; t < T_; ++t) {
    size_t o = ((size_t)b * T_ + t) * L_ + l;
    so += zo[o]; ss += zs[o]; st += zt[o];
  }
  float invT = 1.0f / T_;
  feat[(size_t)b * 193 + l]       = so * invT;
  feat[(size_t)b * 193 + 64 + l]  = ss * invT;
  feat[(size_t)b * 193 + 128 + l] = st * invT;
  float msum = 0.f;
  for (int i = l; i < T_ * F_; i += 64) msum += M[(size_t)b * T_ * F_ + i];
  sm[l] = msum;
  __syncthreads();
  if (l == 0) {
    float s = 0.f;
    for (int i = 0; i < 64; ++i) s += sm[i];
    feat[(size_t)b * 193 + 192] = 1.0f - s / (float)(T_ * F_);
  }
}

// act: 0=none, 1=gauss (1 - exp(-x^2))
__global__ void aecs_small_dense(const float* __restrict__ in, int Kin,
                                 const float* __restrict__ W,
                                 const float* __restrict__ bias,
                                 float* __restrict__ out, int Nout, int act) {
  int b = blockIdx.x, nI = threadIdx.x;
  float s = bias[nI];
  for (int kI = 0; kI < Kin; ++kI)
    s += in[(size_t)b * Kin + kI] * W[(size_t)kI * Nout + nI];
  if (act == 1) s = 1.0f - __expf(-s * s);
  out[(size_t)b * Nout + nI] = s;
}

__global__ void aecs_softmax3(const float* __restrict__ logits,
                              float* __restrict__ alpha) {
  int b = blockIdx.x * blockDim.x + threadIdx.x;
  if (b >= B_) return;
  float a0 = logits[b * 3], a1 = logits[b * 3 + 1], a2 = logits[b * 3 + 2];
  float mx = fmaxf(a0, fmaxf(a1, a2));
  float e0 = __expf(a0 - mx), e1 = __expf(a1 - mx), e2 = __expf(a2 - mx);
  float inv = 1.0f / (e0 + e1 + e2);
  alpha[b * 3] = e0 * inv; alpha[b * 3 + 1] = e1 * inv; alpha[b * 3 + 2] = e2 * inv;
}

__global__ void aecs_mix(const float* __restrict__ alpha,
                         const float* __restrict__ zo,
                         const float* __restrict__ zs,
                         const float* __restrict__ zt,
                         __bf16* __restrict__ zf) {
  int row = blockIdx.x;  // b*T + t
  int b = row / T_;
  int l = threadIdx.x;   // L_
  size_t o = (size_t)row * L_ + l;
  float v = alpha[b * 3] * zo[o] + alpha[b * 3 + 1] * zs[o] +
            alpha[b * 3 + 2] * zt[o];
  zf[o] = (__bf16)v;
}

// ---------------------------------------------------------------------------
// Host orchestration
// ---------------------------------------------------------------------------
extern "C" void kernel_launch(void* const* d_in, const int* in_sizes, int n_in,
                              void* d_out, int out_size, void* d_ws,
                              size_t ws_size, hipStream_t stream) {
  (void)in_sizes; (void)n_in; (void)out_size; (void)ws_size;
  auto Fp = [&](int i) -> const float* { return (const float*)d_in[i]; };
  const float* X = Fp(0);
  const float* Mask = Fp(1);
  // d_in[2] = k (always 5, baked into KNN_)

  // Flattened param order (jax sorted-key tree flatten):
  // stack layout: bn1(beta,gamma,mean,var) bn2(..) dense(W,b)
  //               lstm1(Wh,Wx,b) lstm2(Wh,Wx,b)
  enum { S_BN1_BETA = 0, S_BN1_GAMMA, S_BN1_MEAN, S_BN1_VAR,
         S_BN2_BETA, S_BN2_GAMMA, S_BN2_MEAN, S_BN2_VAR,
         S_DENSE_W, S_DENSE_B, S_L1_WH, S_L1_WX, S_L1_B,
         S_L2_WH, S_L2_WX, S_L2_B };
  const int IDX_DEC = 3, IDX_EO = 19, IDX_ES = 35, IDX_ET = 51, IDX_GATE = 67;

  // Workspace bump allocator (256B aligned)
  char* ws = (char*)d_ws;
  size_t off = 0;
  auto alloc = [&](size_t bytes) -> void* {
    void* p = ws + off;
    off = (off + bytes + 255) & ~(size_t)255;
    return p;
  };

  const size_t BTF = (size_t)B_ * T_ * F_;
  const int Mrows = B_ * T_;

  __bf16* AMr  = (__bf16*)alloc(BTF * 2);
  __bf16* A2Mr = (__bf16*)alloc(BTF * 2);
  __bf16* Mr   = (__bf16*)alloc(BTF * 2);
  __bf16* AMt  = (__bf16*)alloc(BTF * 2);
  __bf16* A2Mt = (__bf16*)alloc(BTF * 2);
  __bf16* Mt   = (__bf16*)alloc(BTF * 2);
  float*  dbuf = (float*)alloc((size_t)B_ * T_ * T_ * 4);
  int*    idxb = (int*)alloc((size_t)B_ * T_ * KNN_ * 4);
  float*  wb   = (float*)alloc((size_t)B_ * T_ * KNN_ * 4);
  float*  Xs   = (float*)alloc(BTF * 4);
  float*  Xt   = (float*)alloc(BTF * 4);
  __bf16* Ain  = (__bf16*)alloc((size_t)Mrows * 2 * F_ * 2);
  float*  xg   = (float*)alloc((size_t)Mrows * G4_ * 4);
  float*  hsb  = (float*)alloc((size_t)Mrows * H_ * 4);
  __bf16* hbn  = (__bf16*)alloc((size_t)Mrows * H_ * 2);
  float*  z_o  = (float*)alloc((size_t)Mrows * L_ * 4);
  float*  z_s  = (float*)alloc((size_t)Mrows * L_ * 4);
  float*  z_t  = (float*)alloc((size_t)Mrows * L_ * 4);
  __bf16* zf_bf = (__bf16*)alloc((size_t)Mrows * L_ * 2);
  float*  feat = (float*)alloc((size_t)B_ * 193 * 4);
  float*  g1   = (float*)alloc((size_t)B_ * 2 * L_ * 4);
  float*  g2   = (float*)alloc((size_t)B_ * L_ * 4);
  float*  logits = (float*)alloc((size_t)B_ * 3 * 4);
  float*  alpha  = (float*)alloc((size_t)B_ * 3 * 4);

  struct StackW { __bf16 *wx1T, *wh1T, *wx2T, *wh2T, *dWT; };
  auto wtrans = [&](const float* W, __bf16* WT, int K, int N) {
    int tot = K * N;
    aecs_wtrans<<<(tot + 255) / 256, 256, 0, stream>>>(W, WT, K, N);
  };
  auto prep_stack = [&](int base, int Kin1, int denseOut) -> StackW {
    StackW s;
    s.wx1T = (__bf16*)alloc((size_t)G4_ * Kin1 * 2);
    s.wh1T = (__bf16*)alloc((size_t)G4_ * H_ * 2);
    s.wx2T = (__bf16*)alloc((size_t)G4_ * H_ * 2);
    s.wh2T = (__bf16*)alloc((size_t)G4_ * H_ * 2);
    s.dWT  = (__bf16*)alloc((size_t)denseOut * H_ * 2);
    wtrans(Fp(base + S_L1_WX), s.wx1T, Kin1, G4_);
    wtrans(Fp(base + S_L1_WH), s.wh1T, H_, G4_);
    wtrans(Fp(base + S_L2_WX), s.wx2T, H_, G4_);
    wtrans(Fp(base + S_L2_WH), s.wh2T, H_, G4_);
    wtrans(Fp(base + S_DENSE_W), s.dWT, H_, denseOut);
    return s;
  };

  const int LSTM_SHMEM = G4_ * H_ * 2 + 16 * H_ * 2;  // 132 KB (CDNA5 LDS)
  hipFuncSetAttribute((const void*)aecs_lstm_rec,
                      hipFuncAttributeMaxDynamicSharedMemorySize, LSTM_SHMEM);

  StackW swDec = prep_stack(IDX_DEC, L_, F_);
  StackW swEO  = prep_stack(IDX_EO, 2 * F_, L_);
  StackW swES  = prep_stack(IDX_ES, 2 * F_, L_);
  StackW swET  = prep_stack(IDX_ET, 2 * F_, L_);

  // ---- KNN imputation ----
  aecs_prep_pdist<<<B_ * T_, F_, 0, stream>>>(X, Mask, AMr, A2Mr, Mr, AMt, A2Mt, Mt);
  // spatial (contract over F)
  aecs_pdist<<<dim3(T_ / 16, T_ / 16, B_), 32, 0, stream>>>(AMr, A2Mr, Mr, T_, F_, dbuf);
  aecs_knn_topk<<<(B_ * T_ + 255) / 256, 256, 0, stream>>>(dbuf, T_, B_ * T_, idxb, wb);
  aecs_impute_spatial<<<B_ * T_, F_, 0, stream>>>(X, Mask, idxb, wb, Xs);
  // temporal (contract over T)
  aecs_pdist<<<dim3(F_ / 16, F_ / 16, B_), 32, 0, stream>>>(AMt, A2Mt, Mt, F_, T_, dbuf);
  aecs_knn_topk<<<(B_ * F_ + 255) / 256, 256, 0, stream>>>(dbuf, F_, B_ * F_, idxb, wb);
  aecs_impute_temporal<<<B_ * F_, T_, 0, stream>>>(X, Mask, idxb, wb, Xt);

  // ---- encoder / decoder stacks ----
  auto run_stack = [&](const __bf16* A0, int Kin1, int denseOut, int base,
                       const StackW& sw, float* zout) {
    aecs_gemm_bf16<<<dim3(G4_ / 64, Mrows / 128), 256, 0, stream>>>(
        A0, sw.wx1T, Fp(base + S_L1_B), xg, Mrows, G4_, Kin1);
    aecs_lstm_rec<<<B_ / 16, 256, LSTM_SHMEM, stream>>>(xg, sw.wh1T, hsb);
    aecs_bn_bf16<<<Mrows, H_, 0, stream>>>(hsb, Fp(base + S_BN1_BETA),
        Fp(base + S_BN1_GAMMA), Fp(base + S_BN1_MEAN), Fp(base + S_BN1_VAR), hbn);
    aecs_gemm_bf16<<<dim3(G4_ / 64, Mrows / 128), 256, 0, stream>>>(
        hbn, sw.wx2T, Fp(base + S_L2_B), xg, Mrows, G4_, H_);
    aecs_lstm_rec<<<B_ / 16, 256, LSTM_SHMEM, stream>>>(xg, sw.wh2T, hsb);
    aecs_bn_bf16<<<Mrows, H_, 0, stream>>>(hsb, Fp(base + S_BN2_BETA),
        Fp(base + S_BN2_GAMMA), Fp(base + S_BN2_MEAN), Fp(base + S_BN2_VAR), hbn);
    aecs_gemm_bf16<<<dim3(denseOut / 64, Mrows / 128), 256, 0, stream>>>(
        hbn, sw.dWT, Fp(base + S_DENSE_B), zout, Mrows, denseOut, H_);
  };

  aecs_encin<<<B_ * T_, 2 * F_, 0, stream>>>(X, Mask, Ain);
  run_stack(Ain, 2 * F_, L_, IDX_EO, swEO, z_o);
  aecs_encin<<<B_ * T_, 2 * F_, 0, stream>>>(Xs, Mask, Ain);
  run_stack(Ain, 2 * F_, L_, IDX_ES, swES, z_s);
  aecs_encin<<<B_ * T_, 2 * F_, 0, stream>>>(Xt, Mask, Ain);
  run_stack(Ain, 2 * F_, L_, IDX_ET, swET, z_t);

  // ---- gating ----
  aecs_feat<<<B_, 64, 0, stream>>>(z_o, z_s, z_t, Mask, feat);
  aecs_small_dense<<<B_, 2 * L_, 0, stream>>>(feat, 3 * L_ + 1, Fp(IDX_GATE + 0),
                                              Fp(IDX_GATE + 1), g1, 2 * L_, 1);
  aecs_small_dense<<<B_, L_, 0, stream>>>(g1, 2 * L_, Fp(IDX_GATE + 2),
                                          Fp(IDX_GATE + 3), g2, L_, 1);
  aecs_small_dense<<<B_, 3, 0, stream>>>(g2, L_, Fp(IDX_GATE + 4),
                                         Fp(IDX_GATE + 5), logits, 3, 0);
  aecs_softmax3<<<1, B_, 0, stream>>>(logits, alpha);
  aecs_mix<<<B_ * T_, L_, 0, stream>>>(alpha, z_o, z_s, z_t, zf_bf);

  // ---- decoder (final dense writes d_out: (B,T,F) f32) ----
  run_stack(zf_bf, L_, F_, IDX_DEC, swDec, (float*)d_out);
}
